// VQ_9517647527985
// MI455X (gfx1250) — matrix operand order repaired and checked by reading, not compiled, not run on hardware
//
#include <hip/hip_runtime.h>
#include <hip/hip_bf16.h>

typedef __attribute__((ext_vector_type(16))) __bf16 v16bf;
typedef __attribute__((ext_vector_type(8)))  __bf16 bf16x8;
typedef __attribute__((ext_vector_type(4)))  __bf16 bf16x4;
typedef __attribute__((ext_vector_type(8)))  float  v8f;

#define D_DIM     256
#define K_CODES   1024
#define M_TOTAL   32768          // B*T = 16*2048
#define TILE_M    128
#define TILE_N    64
#define LDS_STRIDE (D_DIM + 8)   // 264 bf16 = 528 B (16B aligned, bank-conflict-free b128 reads)

// ---------------------------------------------------------------------------
// Kernel 1: per-code squared norms |e_k|^2, and zero the loss accumulator.
// grid = K_CODES, block = 256 (== D_DIM)
// ---------------------------------------------------------------------------
__global__ __launch_bounds__(256)
void vq_enorm_kernel(const float* __restrict__ emb,
                     float* __restrict__ enorm,
                     float* __restrict__ lossSlot) {
    __shared__ float red[8];
    const int code = blockIdx.x;
    const float v = emb[code * D_DIM + threadIdx.x];
    float s = v * v;
    for (int m = 16; m >= 1; m >>= 1) s += __shfl_xor(s, m, 32);
    const int wave = threadIdx.x >> 5, lane = threadIdx.x & 31;
    if (lane == 0) red[wave] = s;
    __syncthreads();
    if (threadIdx.x == 0) {
        float t = 0.f;
        for (int w = 0; w < 8; ++w) t += red[w];
        enorm[code] = t;
        if (code == 0) *lossSlot = 0.f;
    }
}

// ---------------------------------------------------------------------------
// Kernel 2: WMMA distance GEMM + per-row argmin over the codebook.
// grid = M_TOTAL/TILE_M = 256, block = 256 (8 waves; wave w owns rows [16w,16w+16))
// score(row, code) = |e|^2 - 2 * <x_row, e_code>   (|x|^2 dropped: row-constant)
// Each wave runs TWO independent 16x16 accumulation chains (two N-subtiles)
// with double-buffered B fragments to overlap ds_load with v_wmma.
// ---------------------------------------------------------------------------
__global__ __launch_bounds__(256)
void vq_argmin_kernel(const float* __restrict__ x,
                      const float* __restrict__ emb,
                      const float* __restrict__ enorm,
                      int*   __restrict__ idxOut,
                      float* __restrict__ idxOutF) {
    __shared__ __bf16 As[TILE_M * LDS_STRIDE];   // 67.5 KB
    __shared__ __bf16 Bs[TILE_N * LDS_STRIDE];   // 33.8 KB
    __shared__ float  En[TILE_N];

    const int tid   = threadIdx.x;
    const int lane  = tid & 31;
    const int wave  = tid >> 5;
    const int mBase = blockIdx.x * TILE_M;

    // ---- Stage A-tile: 128 rows of x, f32 -> bf16, into LDS ----
    {
        const float4* xsrc = (const float4*)(x + (size_t)mBase * D_DIM);
        for (int i = 0; i < (TILE_M * D_DIM / 4) / 256; ++i) {     // 32 iters
            const int e4  = i * 256 + tid;
            const float4 f = xsrc[e4];
            const int e   = e4 * 4;
            const int row = e / D_DIM, col = e % D_DIM;
            bf16x4 h = { (__bf16)f.x, (__bf16)f.y, (__bf16)f.z, (__bf16)f.w };
            *(bf16x4*)(&As[row * LDS_STRIDE + col]) = h;
        }
    }
    __syncthreads();

    // ---- Load this wave's A fragments (whole D=256) into registers ----
    // ISA A layout (16-bit 16x32): lanes 0-15 row M=L hold K={kb..kb+7, kb+16..kb+23}
    // with kb = ks*32; lanes 16-31 same rows with kb = ks*32 + 8.
    v16bf a[8];
    {
        const int row = wave * 16 + (lane & 15);
        const __bf16* arow = &As[row * LDS_STRIDE];
        for (int ks = 0; ks < 8; ++ks) {
            const int kb = ks * 32 + (lane >> 4) * 8;
            union { bf16x8 h[2]; v16bf v; } u;
            u.h[0] = *(const bf16x8*)(arow + kb);
            u.h[1] = *(const bf16x8*)(arow + kb + 16);
            a[ks] = u.v;
        }
    }

    float best[8];
    int   bestIdx[8];
    for (int r = 0; r < 8; ++r) { best[r] = 3.4e38f; bestIdx[r] = 0; }

    const int c     = lane & 15;        // column within a 16-code subtile
    const int khalf = (lane >> 4) * 8;  // B fragment K-offset per lane half

    for (int nt = 0; nt < K_CODES / TILE_N; ++nt) {
        __syncthreads();   // previous tile's consumers done before restaging Bs

        // ---- Stage B-tile: 64 codebook rows, f32 -> bf16, into LDS ----
        {
            const float4* bsrc = (const float4*)(emb + (size_t)nt * TILE_N * D_DIM);
            for (int i = 0; i < (TILE_N * D_DIM / 4) / 256; ++i) { // 16 iters
                const int e4  = i * 256 + tid;
                const float4 f = bsrc[e4];
                const int e   = e4 * 4;
                const int row = e / D_DIM, col = e % D_DIM;
                bf16x4 h = { (__bf16)f.x, (__bf16)f.y, (__bf16)f.z, (__bf16)f.w };
                *(bf16x4*)(&Bs[row * LDS_STRIDE + col]) = h;
            }
            if (tid < TILE_N) En[tid] = enorm[nt * TILE_N + tid];
            if (nt + 1 < K_CODES / TILE_N) {
                // global_prefetch_b8: one cacheline-trigger per 256B per thread
                const char* nxt = (const char*)(emb + (size_t)(nt + 1) * TILE_N * D_DIM);
                __builtin_prefetch(nxt + tid * 256, 0, 1);
            }
        }
        __syncthreads();

        // ---- 2 passes x 2 concurrent 16-code subtiles, K=256 = 8 WMMA steps ----
        #pragma unroll
        for (int nsp = 0; nsp < 2; ++nsp) {
            const __bf16* brow0 = &Bs[(nsp * 32 +      c) * LDS_STRIDE];
            const __bf16* brow1 = &Bs[(nsp * 32 + 16 + c) * LDS_STRIDE];

            union BU { bf16x8 h[2]; v16bf v; };
            BU b0[2], b1[2];
            // preload ks=0 fragments for both subtiles
            b0[0].h[0] = *(const bf16x8*)(brow0 + khalf);
            b0[0].h[1] = *(const bf16x8*)(brow0 + khalf + 16);
            b1[0].h[0] = *(const bf16x8*)(brow1 + khalf);
            b1[0].h[1] = *(const bf16x8*)(brow1 + khalf + 16);

            v8f acc0 = {0.f, 0.f, 0.f, 0.f, 0.f, 0.f, 0.f, 0.f};
            v8f acc1 = {0.f, 0.f, 0.f, 0.f, 0.f, 0.f, 0.f, 0.f};

            #pragma unroll
            for (int ks = 0; ks < 8; ++ks) {
                const int cur = ks & 1, nxt = cur ^ 1;
                if (ks < 7) {   // prefetch next k-step fragments (double buffer)
                    const int kb = (ks + 1) * 32 + khalf;
                    b0[nxt].h[0] = *(const bf16x8*)(brow0 + kb);
                    b0[nxt].h[1] = *(const bf16x8*)(brow0 + kb + 16);
                    b1[nxt].h[0] = *(const bf16x8*)(brow1 + kb);
                    b1[nxt].h[1] = *(const bf16x8*)(brow1 + kb + 16);
                }
                acc0 = __builtin_amdgcn_wmma_f32_16x16x32_bf16(
                         false, a[ks], false, b0[cur].v, (short)0, acc0, false, false);
                acc1 = __builtin_amdgcn_wmma_f32_16x16x32_bf16(
                         false, a[ks], false, b1[cur].v, (short)0, acc1, false, false);
            }

            const int   code0 = nt * TILE_N + nsp * 32 + c;
            const int   code1 = code0 + 16;
            const float en0   = En[nsp * 32 + c];
            const float en1   = En[nsp * 32 + 16 + c];
            // C layout: lane holds column N=lane&15; VGPR r holds row r (+8 for lanes>=16)
            for (int r = 0; r < 8; ++r) {
                const float s0 = en0 - 2.0f * acc0[r];
                const float s1 = en1 - 2.0f * acc1[r];
                if (s0 < best[r]) { best[r] = s0; bestIdx[r] = code0; }
                if (s1 < best[r]) { best[r] = s1; bestIdx[r] = code1; }
            }
        }
    }

    // ---- Cross-lane argmin within each 16-lane half (first-index tie-break) ----
    for (int r = 0; r < 8; ++r) {
        float b  = best[r];
        int   bi = bestIdx[r];
        for (int s = 1; s < 16; s <<= 1) {
            const float ob = __shfl_xor(b, s, 32);
            const int   oi = __shfl_xor(bi, s, 32);
            if (ob < b || (ob == b && oi < bi)) { b = ob; bi = oi; }
        }
        if ((lane & 15) == 0) {
            const int gm = mBase + wave * 16 + (lane >> 4) * 8 + r;
            idxOut[gm]  = bi;
            idxOutF[gm] = (float)bi;
        }
    }
}

// ---------------------------------------------------------------------------
// Kernel 3: gather values = emb[idx], accumulate loss = 2*mean((x-values)^2).
// grid = M_TOTAL/8 = 4096, block = 256 (== D_DIM; thread t handles column t)
// ---------------------------------------------------------------------------
__global__ __launch_bounds__(256)
void vq_gather_loss_kernel(const float* __restrict__ x,
                           const float* __restrict__ emb,
                           const int*   __restrict__ idx,
                           float* __restrict__ values,
                           float* __restrict__ lossSlot) {
    __shared__ float red[8];
    const int tid     = threadIdx.x;
    const int rowBase = blockIdx.x * 8;
    float s = 0.f;
    for (int r = 0; r < 8; ++r) {
        const int row  = rowBase + r;
        const int code = idx[row];
        const float v  = emb[(size_t)code * D_DIM + tid];
        const float xv = x[(size_t)row * D_DIM + tid];
        values[(size_t)row * D_DIM + tid] = v;
        const float d = xv - v;
        s += d * d;
    }
    for (int m = 16; m >= 1; m >>= 1) s += __shfl_xor(s, m, 32);
    const int wave = tid >> 5, lane = tid & 31;
    if (lane == 0) red[wave] = s;
    __syncthreads();
    if (tid == 0) {
        float t = 0.f;
        for (int w = 0; w < 8; ++w) t += red[w];
        atomicAdd(lossSlot, t * (2.0f / (float)((size_t)M_TOTAL * D_DIM)));
    }
}

// ---------------------------------------------------------------------------
extern "C" void kernel_launch(void* const* d_in, const int* in_sizes, int n_in,
                              void* d_out, int out_size, void* d_ws, size_t ws_size,
                              hipStream_t stream) {
    const float* x   = (const float*)d_in[0];   // [16,2048,256] f32
    const float* emb = (const float*)d_in[1];   // [1024,256]    f32

    float* out    = (float*)d_out;
    float* values = out;                                    // [M_TOTAL*D]
    float* idxF   = out + (size_t)M_TOTAL * D_DIM;          // [M_TOTAL] (as float)
    float* loss   = idxF + M_TOTAL;                         // [1]

    float* enorm = (float*)d_ws;                            // K floats
    int*   idxI  = (int*)((char*)d_ws + K_CODES * sizeof(float)); // M_TOTAL ints

    vq_enorm_kernel<<<K_CODES, 256, 0, stream>>>(emb, enorm, loss);
    vq_argmin_kernel<<<M_TOTAL / TILE_M, 256, 0, stream>>>(x, emb, enorm, idxI, idxF);
    vq_gather_loss_kernel<<<M_TOTAL / 8, 256, 0, stream>>>(x, emb, idxI, values, loss);
}
